// NaMMAttention_10110353015208
// MI455X (gfx1250) — compile-verified, matrix operand order checked by reading, not run on hardware
//
#include <hip/hip_runtime.h>
#include <math.h>

typedef __attribute__((ext_vector_type(16))) __bf16 v16bf;
typedef __attribute__((ext_vector_type(8)))  float  v8f;

// Problem constants (from reference)
constexpr int H_   = 12;
constexpr int D_   = 128;
constexpr int DIM_ = 1536;
constexpr int INNER_ = H_ * D_;      // 1536
constexpr int B_   = 2;
constexpr int LV_  = 1920;           // 2*24*40
constexpr int LT_  = 128;
constexpr int L_   = LV_ + LT_;      // 2048

// Async memory->LDS copy of 16 bytes (per lane), tracked by ASYNCcnt.
// ldsoff: 32-bit LDS byte offset; gptr: 64-bit global address.
// The instruction immediate offset is added to BOTH addresses (ISA 10.x).
__device__ __forceinline__ void async_ld_lds_b128(unsigned ldsoff, const void* gptr) {
  asm volatile("global_load_async_to_lds_b128 %0, %1, off"
               :: "v"(ldsoff), "v"(gptr) : "memory");
}
__device__ __forceinline__ void async_ld_lds_b128_o16(unsigned ldsoff, const void* gptr) {
  asm volatile("global_load_async_to_lds_b128 %0, %1, off offset:16"
               :: "v"(ldsoff), "v"(gptr) : "memory");
}
__device__ __forceinline__ void wait_async0() {
  asm volatile("s_wait_asynccnt 0x0" ::: "memory");
}
__device__ __forceinline__ unsigned lds_off(const void* p) {
  return (unsigned)(unsigned long long)p;   // low 32 bits = LDS byte offset
}

// ---------------------------------------------------------------------------
// f32 -> bf16 conversion (grid-stride)
// ---------------------------------------------------------------------------
__global__ __launch_bounds__(256) void f32_to_bf16(const float* __restrict__ src,
                                                   __bf16* __restrict__ dst,
                                                   long long n) {
  long long i = (long long)blockIdx.x * blockDim.x + threadIdx.x;
  long long stride = (long long)gridDim.x * blockDim.x;
  for (; i < n; i += stride) dst[i] = (__bf16)src[i];
}

// ---------------------------------------------------------------------------
// C[M,N] = A[M,K](bf16) @ B[K,N](bf16) + bias, f32 accumulate via WMMA bf16.
// Workgroup tile 128x128, 8 waves (4x2), wave tile 32x64, K-step 32.
// A staged [128][32] via GLOBAL_LOAD_ASYNC_TO_LDS_B128 (no VGPR round trip);
// B staged transposed [128 n][32 k] so every WMMA fragment is one contiguous
// 32B LDS load.
// ---------------------------------------------------------------------------
__global__ __launch_bounds__(256) void gemm_bf16_wmma(
    const __bf16* __restrict__ A, const __bf16* __restrict__ Bw,
    const float* __restrict__ bias, float* __restrict__ C,
    int M, int N, int K) {
  __shared__ __bf16 As[128][32];
  __shared__ __bf16 BsT[128][32];

  const int tid = threadIdx.x;
  const int wid = tid >> 5;
  const int lane = tid & 31;
  const int l16 = lane & 15;
  const int kh  = lane >> 4;

  const int mtiles = M >> 7;
  const int bm = (blockIdx.x % mtiles) << 7;
  const int bn = (blockIdx.x / mtiles) << 7;
  const int wm = (wid >> 1) << 5;   // 0,32,64,96
  const int wn = (wid & 1) << 6;    // 0,64

  const v8f vz = {0.f, 0.f, 0.f, 0.f, 0.f, 0.f, 0.f, 0.f};
  v8f acc[2][4];
#pragma unroll
  for (int i = 0; i < 2; ++i)
#pragma unroll
    for (int j = 0; j < 4; ++j) acc[i][j] = vz;

  const int ar = tid >> 1, ac = (tid & 1) << 4;       // A: row, col chunk
  const int bkr = tid >> 3, bnc = (tid & 7) << 4;     // B: k-row, n chunk
  const unsigned ldsA = lds_off(&As[ar][ac]);

  for (int k0 = 0; k0 < K; k0 += 32) {
    // Stage A: direct memory->LDS async copy (2 x b128 per lane)
    const __bf16* agp = A + (size_t)(bm + ar) * K + k0 + ac;
    async_ld_lds_b128(ldsA, agp);
    async_ld_lds_b128_o16(ldsA, agp);
    // Stage B transposed (coalesced global read, b16 LDS scatter)
    const __bf16* bsrc = Bw + (size_t)(k0 + bkr) * N + bn + bnc;
#pragma unroll
    for (int i = 0; i < 16; ++i) BsT[bnc + i][bkr] = bsrc[i];
    if (k0 + 32 < K) {
      __builtin_prefetch(A + (size_t)(bm + ar) * K + k0 + 32 + ac, 0, 1);
      __builtin_prefetch(Bw + (size_t)(k0 + 32 + bkr) * N + bn + bnc, 0, 1);
    }
    wait_async0();
    __syncthreads();

#pragma unroll
    for (int mt = 0; mt < 2; ++mt) {
      v16bf a = *(const v16bf*)&As[wm + mt * 16 + l16][kh * 16];
#pragma unroll
      for (int nt = 0; nt < 4; ++nt) {
        v16bf b = *(const v16bf*)&BsT[wn + nt * 16 + l16][kh * 16];
        acc[mt][nt] = __builtin_amdgcn_wmma_f32_16x16x32_bf16(
            false, a, false, b, (short)0, acc[mt][nt], false, false);
      }
    }
    __syncthreads();
  }

  // Epilogue: C layout lane j<16 -> (M=r, N=j); lane>=16 -> (M=8+r, N=j-16)
#pragma unroll
  for (int mt = 0; mt < 2; ++mt) {
#pragma unroll
    for (int nt = 0; nt < 4; ++nt) {
      int col = bn + wn + nt * 16 + l16;
      float bv = bias[col];
#pragma unroll
      for (int r = 0; r < 8; ++r) {
        int row = bm + wm + mt * 16 + kh * 8 + r;
        C[(size_t)row * N + col] = acc[mt][nt][r] + bv;
      }
    }
  }
}

// ---------------------------------------------------------------------------
// RMSNorm (q,k) + pack to bf16 q/k/v (B,L,H,D). One wave per (row, s, head).
// ---------------------------------------------------------------------------
__global__ __launch_bounds__(256) void rmsnorm_pack(
    const float* __restrict__ qkv,       // R x (3*H*D), bias already fused
    const float* __restrict__ nq, const float* __restrict__ nk,
    __bf16* __restrict__ q, __bf16* __restrict__ k, __bf16* __restrict__ v,
    int R, int Lloc, int lofs) {
  const int wid = threadIdx.x >> 5, lane = threadIdx.x & 31;
  int idx = blockIdx.x * 8 + wid;
  int total = R * 3 * H_;
  if (idx >= total) return;
  int h = idx % H_;
  int s = (idx / H_) % 3;
  int row = idx / (3 * H_);
  int b = row / Lloc;
  int l = lofs + row % Lloc;

  const float* src = qkv + (size_t)row * (3 * H_ * D_) + (size_t)(s * H_ + h) * D_;
  float x0 = src[lane], x1 = src[lane + 32], x2 = src[lane + 64], x3 = src[lane + 96];
  if (s < 2) {
    float ss = x0 * x0 + x1 * x1 + x2 * x2 + x3 * x3;
#pragma unroll
    for (int off = 16; off > 0; off >>= 1) ss += __shfl_xor(ss, off, 32);
    float inv = rsqrtf(ss * (1.0f / D_) + 1e-6f);
    const float* w = (s == 0) ? nq : nk;
    x0 *= inv * w[lane];      x1 *= inv * w[lane + 32];
    x2 *= inv * w[lane + 64]; x3 *= inv * w[lane + 96];
  }
  __bf16* dstb = (s == 0 ? q : (s == 1 ? k : v));
  __bf16* dst = dstb + (((size_t)b * L_ + l) * H_ + h) * D_;
  dst[lane]      = (__bf16)x0; dst[lane + 32] = (__bf16)x1;
  dst[lane + 64] = (__bf16)x2; dst[lane + 96] = (__bf16)x3;
}

// ---------------------------------------------------------------------------
// Flash attention, bf16 WMMA. One workgroup = one (b,h) and 128 Q rows
// (16 per wave). KV consumed in 32-row chunks; K staged via async-to-LDS,
// V staged transposed; online softmax in f32 on the C-fragment layout;
// P re-layouted via per-wave LDS tile.
// ---------------------------------------------------------------------------
__global__ __launch_bounds__(256) void flash_attn_wmma(
    const __bf16* __restrict__ q, const __bf16* __restrict__ k,
    const __bf16* __restrict__ v,
    __bf16* __restrict__ avid, __bf16* __restrict__ atxt) {
  __shared__ __bf16 Kl[32][128];       // K chunk, row-major
  __shared__ __bf16 Vt[128][32];       // V chunk, transposed (d-major)
  __shared__ __bf16 Pl[8][16][32];     // per-wave P tile

  const int tid = threadIdx.x, wid = tid >> 5, lane = tid & 31;
  const int l16 = lane & 15, kh = lane >> 4;
  const int qt = blockIdx.x & 15;
  const int h  = (blockIdx.x >> 4) % H_;
  const int b  = blockIdx.x / (16 * H_);
  const int qrow = qt * 128 + wid * 16;

  // Q A-fragments: lane l16 = row, kh selects K-halves; 4 k-steps cover D=128
  const __bf16* qp = q + (((size_t)b * L_ + qrow + l16) * H_ + h) * D_;
  v16bf qf[4];
#pragma unroll
  for (int kk = 0; kk < 4; ++kk)
    qf[kk] = *(const v16bf*)(qp + kk * 32 + kh * 16);

  const v8f vz = {0.f, 0.f, 0.f, 0.f, 0.f, 0.f, 0.f, 0.f};
  v8f o[8];
#pragma unroll
  for (int i = 0; i < 8; ++i) o[i] = vz;
  float mrow[8], lrow[8];
#pragma unroll
  for (int r = 0; r < 8; ++r) { mrow[r] = -INFINITY; lrow[r] = 0.f; }

  const float scale = 0.088388347648318447f;   // 1/sqrt(128)
  const int skr = tid >> 3, snc = (tid & 7) << 4;
  const unsigned ldsK = lds_off(&Kl[skr][snc]);

  for (int c0 = 0; c0 < L_; c0 += 32) {
    // Stage K rows: direct memory->LDS async copy
    const __bf16* kg = k + (((size_t)b * L_ + c0 + skr) * H_ + h) * D_ + snc;
    async_ld_lds_b128(ldsK, kg);
    async_ld_lds_b128_o16(ldsK, kg);
    // Stage V transposed (needs element scatter, so classic path)
    const __bf16* vs = v + (((size_t)b * L_ + c0 + skr) * H_ + h) * D_ + snc;
#pragma unroll
    for (int i = 0; i < 16; ++i) Vt[snc + i][skr] = vs[i];
    wait_async0();
    __syncthreads();

    // S = Q * K^T : 16x32 chunk as two 16x16 accumulators
    v8f s0 = vz, s1 = vz;
#pragma unroll
    for (int kk = 0; kk < 4; ++kk) {
      v16bf kb0 = *(const v16bf*)&Kl[l16][kk * 32 + kh * 16];
      s0 = __builtin_amdgcn_wmma_f32_16x16x32_bf16(false, qf[kk], false, kb0,
                                                   (short)0, s0, false, false);
      v16bf kb1 = *(const v16bf*)&Kl[16 + l16][kk * 32 + kh * 16];
      s1 = __builtin_amdgcn_wmma_f32_16x16x32_bf16(false, qf[kk], false, kb1,
                                                   (short)0, s1, false, false);
    }

    // Online softmax on C-layout: lane group kh handles rows kh*8 + r
    float cf[8];
#pragma unroll
    for (int r = 0; r < 8; ++r) {
      float a0 = s0[r] * scale, a1 = s1[r] * scale;
      float mx = fmaxf(a0, a1);
#pragma unroll
      for (int off = 8; off > 0; off >>= 1)
        mx = fmaxf(mx, __shfl_xor(mx, off, 32));
      float mn = fmaxf(mrow[r], mx);
      cf[r] = __expf(mrow[r] - mn);
      mrow[r] = mn;
      float p0 = __expf(a0 - mn);
      float p1 = __expf(a1 - mn);
      Pl[wid][kh * 8 + r][l16]      = (__bf16)p0;
      Pl[wid][kh * 8 + r][16 + l16] = (__bf16)p1;
      float ps = p0 + p1;
#pragma unroll
      for (int off = 8; off > 0; off >>= 1) ps += __shfl_xor(ps, off, 32);
      lrow[r] = lrow[r] * cf[r] + ps;
    }
    // Rescale O accumulators
#pragma unroll
    for (int nt = 0; nt < 8; ++nt)
#pragma unroll
      for (int r = 0; r < 8; ++r) o[nt][r] *= cf[r];

    // Wave-local LDS roundtrip for P: only DS ordering needed
    asm volatile("s_wait_dscnt 0x0" ::: "memory");
    v16bf pf = *(const v16bf*)&Pl[wid][l16][kh * 16];

    // O += P * V : 8 n-tiles of 16 over D=128, K'=32
#pragma unroll
    for (int nt = 0; nt < 8; ++nt) {
      v16bf vf = *(const v16bf*)&Vt[nt * 16 + l16][kh * 16];
      o[nt] = __builtin_amdgcn_wmma_f32_16x16x32_bf16(false, pf, false, vf,
                                                      (short)0, o[nt], false, false);
    }
    __syncthreads();
  }

  // Normalize and scatter to vid/txt attention-output buffers (bf16)
#pragma unroll
  for (int r = 0; r < 8; ++r) {
    int l = qrow + kh * 8 + r;
    float invl = 1.0f / lrow[r];
    __bf16* dst = (l < LV_)
        ? avid + ((size_t)(b * LV_ + l) * (size_t)(H_ * D_) + h * D_)
        : atxt + ((size_t)(b * LT_ + (l - LV_)) * (size_t)(H_ * D_) + h * D_);
#pragma unroll
    for (int nt = 0; nt < 8; ++nt)
      dst[nt * 16 + l16] = (__bf16)(o[nt][r] * invl);
  }
}

// ---------------------------------------------------------------------------
// Host orchestration
// ---------------------------------------------------------------------------
extern "C" void kernel_launch(void* const* d_in, const int* in_sizes, int n_in,
                              void* d_out, int out_size, void* d_ws, size_t ws_size,
                              hipStream_t stream) {
  (void)in_sizes; (void)n_in; (void)out_size; (void)ws_size;

  const float* vid      = (const float*)d_in[0];
  const float* txt      = (const float*)d_in[1];
  const float* Wqkv_vid = (const float*)d_in[2];
  const float* bqkv_vid = (const float*)d_in[3];
  const float* Wqkv_txt = (const float*)d_in[4];
  const float* bqkv_txt = (const float*)d_in[5];
  const float* nq_vid   = (const float*)d_in[6];
  const float* nk_vid   = (const float*)d_in[7];
  const float* nq_txt   = (const float*)d_in[8];
  const float* nk_txt   = (const float*)d_in[9];
  const float* Wout_vid = (const float*)d_in[10];
  const float* bout_vid = (const float*)d_in[11];
  const float* Wout_txt = (const float*)d_in[12];
  const float* bout_txt = (const float*)d_in[13];
  float* outf = (float*)d_out;

  constexpr int MV = B_ * LV_;   // 3840
  constexpr int MT = B_ * LT_;   // 256
  constexpr int N3 = 3 * INNER_; // 4608

  char* ws = (char*)d_ws;
  size_t off = 0;
  auto take = [&](size_t bytes) -> void* {
    void* p = ws + off;
    off = (off + bytes + 255) & ~(size_t)255;
    return p;
  };

  __bf16* vid_b  = (__bf16*)take((size_t)MV * DIM_ * 2);
  __bf16* txt_b  = (__bf16*)take((size_t)MT * DIM_ * 2);
  __bf16* wqv_b  = (__bf16*)take((size_t)DIM_ * N3 * 2);
  __bf16* wqt_b  = (__bf16*)take((size_t)DIM_ * N3 * 2);
  __bf16* wov_b  = (__bf16*)take((size_t)INNER_ * DIM_ * 2);
  __bf16* wot_b  = (__bf16*)take((size_t)INNER_ * DIM_ * 2);
  float*  qkv_v  = (float*)take((size_t)MV * N3 * 4);
  float*  qkv_t  = (float*)take((size_t)MT * N3 * 4);
  __bf16* qb     = (__bf16*)take((size_t)B_ * L_ * INNER_ * 2);
  __bf16* kb     = (__bf16*)take((size_t)B_ * L_ * INNER_ * 2);
  __bf16* vb     = (__bf16*)take((size_t)B_ * L_ * INNER_ * 2);
  __bf16* avid   = (__bf16*)take((size_t)MV * INNER_ * 2);
  __bf16* atxt   = (__bf16*)take((size_t)MT * INNER_ * 2);

  // 1) Convert activations + weights to bf16
  f32_to_bf16<<<2048, 256, 0, stream>>>(vid, vid_b, (long long)MV * DIM_);
  f32_to_bf16<<<512,  256, 0, stream>>>(txt, txt_b, (long long)MT * DIM_);
  f32_to_bf16<<<2048, 256, 0, stream>>>(Wqkv_vid, wqv_b, (long long)DIM_ * N3);
  f32_to_bf16<<<2048, 256, 0, stream>>>(Wqkv_txt, wqt_b, (long long)DIM_ * N3);
  f32_to_bf16<<<1024, 256, 0, stream>>>(Wout_vid, wov_b, (long long)INNER_ * DIM_);
  f32_to_bf16<<<1024, 256, 0, stream>>>(Wout_txt, wot_b, (long long)INNER_ * DIM_);

  // 2) QKV projections (bias fused)
  gemm_bf16_wmma<<<(MV / 128) * (N3 / 128), 256, 0, stream>>>(
      vid_b, wqv_b, bqkv_vid, qkv_v, MV, N3, DIM_);
  gemm_bf16_wmma<<<(MT / 128) * (N3 / 128), 256, 0, stream>>>(
      txt_b, wqt_b, bqkv_txt, qkv_t, MT, N3, DIM_);

  // 3) RMSNorm q,k + pack to (B,L,H,D) bf16
  rmsnorm_pack<<<(MV * 3 * H_) / 8, 256, 0, stream>>>(
      qkv_v, nq_vid, nk_vid, qb, kb, vb, MV, LV_, 0);
  rmsnorm_pack<<<(MT * 3 * H_) / 8, 256, 0, stream>>>(
      qkv_t, nq_txt, nk_txt, qb, kb, vb, MT, LT_, LV_);

  // 4) Flash attention
  flash_attn_wmma<<<B_ * H_ * (L_ / 128), 256, 0, stream>>>(qb, kb, vb, avid, atxt);

  // 5) Output projections straight into d_out (vid block then txt block)
  gemm_bf16_wmma<<<(MV / 128) * (DIM_ / 128), 256, 0, stream>>>(
      avid, wov_b, bout_vid, outf, MV, DIM_, INNER_);
  gemm_bf16_wmma<<<(MT / 128) * (DIM_ / 128), 256, 0, stream>>>(
      atxt, wot_b, bout_txt, outf + (size_t)MV * DIM_, MT, DIM_, INNER_);
}
